// GraphAttentionPooling_17583596110568
// MI455X (gfx1250) — matrix operand order repaired and checked by reading, not compiled, not run on hardware
//
#include <hip/hip_runtime.h>
#include <math.h>

#define NGRAPH 1024
#define DIM    256
#define HID    128

typedef __attribute__((ext_vector_type(16))) _Float16 v16h;
typedef __attribute__((ext_vector_type(8)))  float    v8f;
typedef __attribute__((ext_vector_type(4)))  float    v4f;
typedef __attribute__((ext_vector_type(4)))  int      v4i;

#if __has_builtin(__builtin_amdgcn_global_load_async_to_lds_b128) && \
    __has_builtin(__builtin_amdgcn_s_wait_asynccnt)
#define USE_ASYNC_LDS 1
#else
#define USE_ASYNC_LDS 0
#endif

// ---------------------------------------------------------------------------
// Kernel 0: W1 [D,H] f32 row-major  ->  W1t [H,D] f16 row-major (B is K-major)
// ---------------------------------------------------------------------------
__global__ void prep_w1t(const float* __restrict__ W1, _Float16* __restrict__ w1t) {
  int idx = blockIdx.x * blockDim.x + threadIdx.x;
  if (idx < HID * DIM) {
    int h = idx >> 8;        // / DIM
    int d = idx & (DIM - 1); // % DIM
    w1t[idx] = (_Float16)W1[d * HID + h];
  }
}

// ---------------------------------------------------------------------------
// Segment starts via binary search (batch is sorted). starts[NGRAPH] = N.
// ---------------------------------------------------------------------------
__global__ void seg_starts(const int* __restrict__ batch, int* __restrict__ starts, int N) {
  int g = blockIdx.x * blockDim.x + threadIdx.x;
  if (g > NGRAPH) return;
  if (g == NGRAPH) { starts[g] = N; return; }
  int lo = 0, hi = N;
  while (lo < hi) {
    int mid = (lo + hi) >> 1;
    if (batch[mid] < g) lo = mid + 1; else hi = mid;
  }
  starts[g] = lo;
}

// ---------------------------------------------------------------------------
// Kernel 1: scores[n] = W2 . relu(x[n] @ W1 + b1) + b2 via f16 WMMA.
// Block = 256 threads = 8 waves; each wave owns a 16-node tile (128/block).
// W1t (64KB f16) staged in LDS once per block (async-to-LDS when available).
// ---------------------------------------------------------------------------
__global__ __launch_bounds__(256) void score_kernel(
    const float* __restrict__ x, const _Float16* __restrict__ w1t,
    const float* __restrict__ b1, const float* __restrict__ W2,
    const float* __restrict__ b2, float* __restrict__ scores, int N)
{
  __shared__ alignas(32) _Float16 ldsW[HID * DIM];   // 64 KB

#if USE_ASYNC_LDS
  {
    char* src = (char*)w1t;           // builtin wants non-const
    char* dst = (char*)ldsW;
    #pragma unroll 4
    for (int i = threadIdx.x; i < (HID * DIM * 2) / 16; i += blockDim.x) {
      __builtin_amdgcn_global_load_async_to_lds_b128(
          (__attribute__((address_space(1))) v4i*)(src + i * 16),
          (__attribute__((address_space(3))) v4i*)(dst + i * 16),
          0, 0);
    }
    __builtin_amdgcn_s_wait_asynccnt(0);
  }
#else
  {
    const uint4* src = (const uint4*)w1t;
    uint4*       dst = (uint4*)ldsW;
    #pragma unroll 4
    for (int i = threadIdx.x; i < (HID * DIM * 2) / 16; i += blockDim.x)
      dst[i] = src[i];
  }
#endif
  __syncthreads();

  const int lane = threadIdx.x & 31;
  const int wave = threadIdx.x >> 5;
  const int row  = lane & 15;   // A-matrix M index / B-matrix N index
  const int hs   = lane >> 4;   // half-wave select (K grouping)

  const int tileBase = blockIdx.x * 128 + wave * 16;
  int node = tileBase + row;
  if (node > N - 1) node = N - 1;          // clamp tail; clamped rows never written
  const float* xr = x + (size_t)node * DIM;

  v8f acc[8] = {};   // 8 N-tiles of 16 h-columns each -> h[16][128]

  #pragma unroll
  for (int ks = 0; ks < 8; ++ks) {         // K chunks of 32 over D=256
    // ---- A fragment: row `row`, K = kb..kb+15 (16 consecutive K per lane, matches B)
    const int kb = ks * 32 + hs * 16;
    v4f f0 = __builtin_nontemporal_load((const v4f*)(xr + kb));
    v4f f1 = __builtin_nontemporal_load((const v4f*)(xr + kb + 4));
    v4f f2 = __builtin_nontemporal_load((const v4f*)(xr + kb + 8));
    v4f f3 = __builtin_nontemporal_load((const v4f*)(xr + kb + 12));
    v16h a;
    a[0]=(_Float16)f0[0];  a[1]=(_Float16)f0[1];  a[2]=(_Float16)f0[2];  a[3]=(_Float16)f0[3];
    a[4]=(_Float16)f1[0];  a[5]=(_Float16)f1[1];  a[6]=(_Float16)f1[2];  a[7]=(_Float16)f1[3];
    a[8]=(_Float16)f2[0];  a[9]=(_Float16)f2[1];  a[10]=(_Float16)f2[2]; a[11]=(_Float16)f2[3];
    a[12]=(_Float16)f3[0]; a[13]=(_Float16)f3[1]; a[14]=(_Float16)f3[2]; a[15]=(_Float16)f3[3];

    const int kb2 = ks * 32 + hs * 16;     // B: 16 consecutive K per lane
    #pragma unroll
    for (int nt = 0; nt < 8; ++nt) {
      const int col = nt * 16 + row;       // h column for this lane
      const v16h b = *(const v16h*)&ldsW[col * DIM + kb2];  // 2x ds_load_b128
      acc[nt] = __builtin_amdgcn_wmma_f32_16x16x32_f16(
          false, a, false, b, (short)0, acc[nt], false, false);
    }
  }

  // Epilogue: relu(h + b1) . W2, reduced over the 128 h-columns.
  // C layout: acc[nt][i] = h[M = i + 8*hs][Ncol = nt*16 + row]
  float p[8];
  #pragma unroll
  for (int i = 0; i < 8; ++i) p[i] = 0.f;

  #pragma unroll
  for (int nt = 0; nt < 8; ++nt) {
    const int col = nt * 16 + row;
    const float bb = b1[col];
    const float ww = W2[col];
    #pragma unroll
    for (int i = 0; i < 8; ++i) {
      float hv = acc[nt][i] + bb;
      hv = fmaxf(hv, 0.f);
      p[i] += hv * ww;
    }
  }

  // reduce across the 16 lanes of each half-wave (columns of the same row)
  #pragma unroll
  for (int mask = 8; mask >= 1; mask >>= 1) {
    #pragma unroll
    for (int i = 0; i < 8; ++i) p[i] += __shfl_xor(p[i], mask, 32);
  }

  if ((lane & 15) == 0) {
    const float bias2 = b2[0];
    const int rbase = hs * 8;
    #pragma unroll
    for (int i = 0; i < 8; ++i) {
      const int nd = tileBase + rbase + i;
      if (nd < N) scores[nd] = p[i] + bias2;
    }
  }
}

// ---------------------------------------------------------------------------
// Online softmax reduction: (max, sum-of-exp) pairs
// ---------------------------------------------------------------------------
__device__ inline void ms_combine(float& m1, float& s1, float m2, float s2) {
  float nm = fmaxf(m1, m2);
  if (nm == -__builtin_inff()) { m1 = nm; s1 = 0.f; return; }
  s1 = s1 * expf(m1 - nm) + s2 * expf(m2 - nm);
  m1 = nm;
}

__global__ __launch_bounds__(256) void softmax_part(
    const float* __restrict__ scores, int N,
    float* __restrict__ pm, float* __restrict__ ps)
{
  const int tid = threadIdx.x;
  const int stride = gridDim.x * blockDim.x;
  float m = -__builtin_inff(), s = 0.f;
  for (int i = blockIdx.x * blockDim.x + tid; i < N; i += stride) {
    const float v = scores[i];
    if (v > m) { s = s * expf(m - v) + 1.f; m = v; }
    else       { s += expf(v - m); }
  }
  __shared__ float sm[256], ss[256];
  sm[tid] = m; ss[tid] = s;
  __syncthreads();
  for (int off = 128; off > 0; off >>= 1) {
    if (tid < off) {
      float m1 = sm[tid], s1 = ss[tid];
      ms_combine(m1, s1, sm[tid + off], ss[tid + off]);
      sm[tid] = m1; ss[tid] = s1;
    }
    __syncthreads();
  }
  if (tid == 0) { pm[blockIdx.x] = sm[0]; ps[blockIdx.x] = ss[0]; }
}

__global__ __launch_bounds__(512) void softmax_final(
    const float* __restrict__ pm, const float* __restrict__ ps,
    int B, float* __restrict__ stats)
{
  const int tid = threadIdx.x;
  float m = (tid < B) ? pm[tid] : -__builtin_inff();
  float s = (tid < B) ? ps[tid] : 0.f;
  __shared__ float sm[512], ss[512];
  sm[tid] = m; ss[tid] = s;
  __syncthreads();
  for (int off = 256; off > 0; off >>= 1) {
    if (tid < off) {
      float m1 = sm[tid], s1 = ss[tid];
      ms_combine(m1, s1, sm[tid + off], ss[tid + off]);
      sm[tid] = m1; ss[tid] = s1;
    }
    __syncthreads();
  }
  if (tid == 0) { stats[0] = sm[0]; stats[1] = ss[0]; stats[2] = 1.f / ss[0]; }
}

// ---------------------------------------------------------------------------
// scores -> attn in place: attn[i] = exp(scores[i]-M)/S  (one cheap 2MB pass,
// removes 256x-redundant v_exp_f32 from the memory-bound pooling pass)
// ---------------------------------------------------------------------------
__global__ __launch_bounds__(256) void attn_kernel(
    float* __restrict__ scores, int N, const float* __restrict__ stats)
{
  const int i = blockIdx.x * blockDim.x + threadIdx.x;
  if (i < N) scores[i] = expf(scores[i] - stats[0]) * stats[2];
}

// ---------------------------------------------------------------------------
// Kernel 3: fused pooling. Block = graph, thread = feature column (coalesced).
// One streaming NT pass over x; x4 unrolled to keep 4 loads in flight/thread.
// Accumulation order matches the scalar loop -> deterministic.
// ---------------------------------------------------------------------------
__global__ __launch_bounds__(256) void pool_kernel(
    const float* __restrict__ x, const float* __restrict__ attn,
    const int* __restrict__ starts, float* __restrict__ out)
{
  const int g = blockIdx.x;
  const int t = threadIdx.x;                // 0..255 feature column
  const int s0 = starts[g], s1 = starts[g + 1];

  float asum = 0.f, vmax = -__builtin_inff(), vsum = 0.f;
  int n = s0;
  for (; n + 4 <= s1; n += 4) {
    const float a0 = attn[n], a1 = attn[n + 1], a2 = attn[n + 2], a3 = attn[n + 3];
    const float* p = x + (size_t)n * DIM + t;
    const float v0 = __builtin_nontemporal_load(p);
    const float v1 = __builtin_nontemporal_load(p + DIM);
    const float v2 = __builtin_nontemporal_load(p + 2 * DIM);
    const float v3 = __builtin_nontemporal_load(p + 3 * DIM);
    asum = fmaf(v0, a0, asum); vsum += v0; vmax = fmaxf(vmax, v0);
    asum = fmaf(v1, a1, asum); vsum += v1; vmax = fmaxf(vmax, v1);
    asum = fmaf(v2, a2, asum); vsum += v2; vmax = fmaxf(vmax, v2);
    asum = fmaf(v3, a3, asum); vsum += v3; vmax = fmaxf(vmax, v3);
  }
  for (; n < s1; ++n) {
    const float a = attn[n];
    const float v = __builtin_nontemporal_load(x + (size_t)n * DIM + t);
    asum = fmaf(v, a, asum); vsum += v; vmax = fmaxf(vmax, v);
  }

  float cnt = (float)(s1 - s0);
  cnt = fmaxf(cnt, 1.f);

  float* og = out + (size_t)g * (3 * DIM);
  og[t]           = asum;        // attn_pooled
  og[DIM + t]     = vmax;        // max_pooled
  og[2 * DIM + t] = vsum / cnt;  // mean_pooled
}

// ---------------------------------------------------------------------------
extern "C" void kernel_launch(void* const* d_in, const int* in_sizes, int n_in,
                              void* d_out, int out_size, void* d_ws, size_t ws_size,
                              hipStream_t stream) {
  const float* x     = (const float*)d_in[0];
  const float* W1    = (const float*)d_in[1];
  const float* b1    = (const float*)d_in[2];
  const float* W2    = (const float*)d_in[3];
  const float* b2    = (const float*)d_in[4];
  const int*   batch = (const int*)d_in[5];
  const int N = in_sizes[0] / DIM;

  // workspace carve-out (~2.1 MB)
  char* ws = (char*)d_ws;
  size_t off = 0;
  float* scores = (float*)(ws + off); off += (size_t)N * 4;          off = (off + 255) & ~(size_t)255;
  _Float16* w1t = (_Float16*)(ws + off); off += HID * DIM * 2;       off = (off + 255) & ~(size_t)255;
  float* pm     = (float*)(ws + off); off += 512 * 4;                off = (off + 255) & ~(size_t)255;
  float* ps     = (float*)(ws + off); off += 512 * 4;                off = (off + 255) & ~(size_t)255;
  float* stats  = (float*)(ws + off); off += 256;
  int*   starts = (int*)(ws + off);

  prep_w1t<<<(HID * DIM + 255) / 256, 256, 0, stream>>>(W1, w1t);
  seg_starts<<<(NGRAPH + 1 + 255) / 256, 256, 0, stream>>>(batch, starts, N);

  const int nScoreBlocks = (N + 127) / 128;   // 128 nodes (8 wave-tiles) per block
  score_kernel<<<nScoreBlocks, 256, 0, stream>>>(x, w1t, b1, W2, b2, scores, N);

  softmax_part<<<512, 256, 0, stream>>>(scores, N, pm, ps);
  softmax_final<<<1, 512, 0, stream>>>(pm, ps, 512, stats);
  attn_kernel<<<(N + 255) / 256, 256, 0, stream>>>(scores, N, stats);

  pool_kernel<<<NGRAPH, 256, 0, stream>>>(x, scores, starts, (float*)d_out);
}
